// CrystalGraphE3ConvNet_89816356094339
// MI455X (gfx1250) — matrix operand order repaired and checked by reading, not compiled, not run on hardware
//
#include <hip/hip_runtime.h>
#include <hip/hip_bf16.h>

// ---------------- problem constants (match reference) ----------------
#define NATOMS 100000
#define M_NBR  12
#define ORIGF  92
#define NBRF   41
#define FDIM   64
#define HFDIM  128
#define NCONV  3
#define BCRY   500
#define NPC    200
#define EDGES  (NATOMS * M_NBR)
#define NB_TP  ((NATOMS + 63) / 64)   // tp-kernel blocks = 1563

#ifndef CG_USE_ASYNC
#define CG_USE_ASYNC 1     // gfx1250 async global->LDS staging
#endif

__device__ __constant__ const float kY0     = 0.28209479177387814f; // 1/(2*sqrt(pi))
__device__ __constant__ const float kTPNORM = 0.125f;               // 1/sqrt(64)
__device__ __constant__ const float kEPS    = 1e-5f;

// ---------------- WMMA plumbing (CDNA5 gfx1250, wave32) ----------------
typedef __attribute__((ext_vector_type(16))) __bf16 v16bf;
typedef __attribute__((ext_vector_type(8)))  float  v8f;

#define DEVINL __device__ __forceinline__

DEVINL float sp_f(float v) {                 // softplus, overflow-safe
    return (v > 20.0f) ? v : log1pf(expf(v));
}

DEVINL v8f wmma_bf16(v16bf a, v16bf b, v8f c) {
    // (neg_a, A, neg_b, B, c_mod, C, reuse_a, reuse_b)
    return __builtin_amdgcn_wmma_f32_16x16x32_bf16(false, a, false, b, (short)0, c,
                                                   false, false);
}

// A tile 16xK (row major in LDS, stride ldA). ISA layout: lane&15 = M,
// lane>>4 selects K sub-block; VGPR v<4 -> K=2v(+8h), v>=4 -> K=16+2(v-4)(+8h).
DEVINL v16bf pack_A(const __bf16* As, int ldA, int lane, int kbase) {
    int m = lane & 15, h = lane >> 4;
    const __bf16* row = As + m * ldA + kbase + h * 8;
    v16bf a;
#pragma unroll
    for (int g = 0; g < 2; ++g)
#pragma unroll
        for (int j = 0; j < 8; ++j)
            a[g * 8 + (j >> 1) * 2 + (j & 1)] = row[g * 16 + j];
    return a;
}

// B tile Kx16 (row major in LDS, stride ldB, pointer pre-offset to n0).
// ISA layout: lane&15 = N, elem e -> K = 8h + 16*(e>>3) + (e&7), h = lane>>4.
DEVINL v16bf pack_B(const __bf16* Bs, int ldB, int lane, int kbase) {
    const __bf16* col = Bs + (lane & 15) + (kbase + (lane >> 4) * 8) * ldB;
    v16bf b;
#pragma unroll
    for (int e = 0; e < 16; ++e) b[e] = col[((e >> 3) * 16 + (e & 7)) * ldB];
    return b;
}

// ---------------- async global->LDS staging (ASYNCcnt-tracked) ----------------
// Copies nfloat4 * 16 bytes from src (16B aligned) into dstlds (16B aligned).
// Caller must __syncthreads() afterwards before reading LDS.
DEVINL void stage_async(float* dstlds, const float* src, int nfloat4, int tid,
                        int nthr) {
#if CG_USE_ASYNC
    unsigned           base = (unsigned)(uintptr_t)dstlds;   // low 32 bits give the LDS address
    unsigned long long g    = (unsigned long long)(uintptr_t)src;
    for (int p = tid; p < nfloat4; p += nthr) {
        unsigned           loff = base + (unsigned)p * 16u;
        unsigned long long ga   = g + (unsigned long long)p * 16u;
        asm volatile("global_load_async_to_lds_b128 %0, %1, off"
                     :: "v"(loff), "v"(ga)
                     : "memory");
    }
    asm volatile("s_wait_asynccnt 0x0" ::: "memory");
#else
    float4*       d  = (float4*)dstlds;
    const float4* sg = (const float4*)src;
    for (int p = tid; p < nfloat4; p += nthr) d[p] = sg[p];
#endif
}

// ---------------- K1: embedding  x = atom_fea @ emb_W + emb_b ----------------
__global__ __launch_bounds__(128) void cg_embed_kernel(
    const float* __restrict__ atom_fea, const float* __restrict__ emb_W,
    const float* __restrict__ emb_b, float* __restrict__ x) {
    __shared__ float  Araw[64 * ORIGF];  // raw f32 tile, async-staged
    __shared__ __bf16 As[64][96];        // K padded 92->96
    __shared__ __bf16 Bs[96][64];
    int tid  = threadIdx.x;
    int row0 = blockIdx.x * 64;

    {   // async stage A with per-packet guard against end of atom_fea allocation
        const float* gsrc  = atom_fea + (long)row0 * ORIGF;
        const int    npkt  = (64 * ORIGF) / 4;               // 1472
        const long   limit = (long)(NATOMS - row0) * ORIGF;  // valid floats
#if CG_USE_ASYNC
        unsigned           base = (unsigned)(uintptr_t)&Araw[0];
        unsigned long long g    = (unsigned long long)(uintptr_t)gsrc;
#endif
        for (int p = tid; p < npkt; p += 128) {
            int fb = p * 4;
            if ((long)fb + 4 <= limit) {
#if CG_USE_ASYNC
                unsigned           loff = base + (unsigned)fb * 4u;
                unsigned long long ga   = g + (unsigned long long)fb * 4u;
                asm volatile("global_load_async_to_lds_b128 %0, %1, off"
                             :: "v"(loff), "v"(ga)
                             : "memory");
#else
                *(float4*)&Araw[fb] = *(const float4*)&gsrc[fb];
#endif
            } else {
                Araw[fb] = 0.f; Araw[fb + 1] = 0.f;
                Araw[fb + 2] = 0.f; Araw[fb + 3] = 0.f;
            }
        }
#if CG_USE_ASYNC
        asm volatile("s_wait_asynccnt 0x0" ::: "memory");
#endif
    }
    for (int i = tid; i < 96 * 64; i += 128) {   // weights, sync stage
        int k = i / 64, n = i % 64;
        Bs[k][n] = (__bf16)((k < ORIGF) ? emb_W[k * FDIM + n] : 0.0f);
    }
    __syncthreads();
    for (int i = tid; i < 64 * 96; i += 128) {   // f32 -> bf16 convert
        int r = i / 96, k = i % 96;
        As[r][k] = (__bf16)((k < ORIGF) ? Araw[r * ORIGF + k] : 0.0f);
    }
    __syncthreads();

    int wave = tid >> 5, lane = tid & 31;
    v8f c[4] = {};
#pragma unroll
    for (int ks = 0; ks < 3; ++ks) {
        v16bf a = pack_A(&As[wave * 16][0], 96, lane, ks * 32);
#pragma unroll
        for (int nt = 0; nt < 4; ++nt) {
            v16bf b = pack_B(&Bs[0][nt * 16], 64, lane, ks * 32);
            c[nt]   = wmma_bf16(a, b, c[nt]);
        }
    }
    int h = lane >> 4, nn = lane & 15;
#pragma unroll
    for (int nt = 0; nt < 4; ++nt) {
        float bias = emb_b[nt * 16 + nn];
#pragma unroll
        for (int r = 0; r < 8; ++r) {
            int gm = row0 + wave * 16 + r + 8 * h;
            if (gm < NATOMS) x[(long)gm * FDIM + nt * 16 + nn] = c[nt][r] + bias;
        }
    }
}

// ---------------- K2: fused edge MLP for all 3 layers -> w0[l][edge] ----------------
__global__ __launch_bounds__(128) void cg_edge_kernel(
    const float* __restrict__ nbr_fea, const float* __restrict__ rW1,
    const float* __restrict__ rb1, const float* __restrict__ rW2,
    const float* __restrict__ rb2, float* __restrict__ w0out) {
    __shared__ float  Araw[64 * NBRF];  // raw f32 edge-feature tile (async)
    __shared__ __bf16 As[64][64];       // 64 edges x K (41 -> 64)
    __shared__ __bf16 Bs[64][48];       // 41x41 -> 64x48
    __shared__ float  hR[64][48];
    __shared__ float  w2col[48];
    int  tid  = threadIdx.x;
    long row0 = (long)blockIdx.x * 64;

    // EDGES % 64 == 0: full tile, no guard. 64*41*4 bytes is 16B-aligned.
    stage_async(&Araw[0], nbr_fea + row0 * NBRF, (64 * NBRF) / 4, tid, 128);
    __syncthreads();
    for (int i = tid; i < 64 * 64; i += 128) {
        int r = i >> 6, k = i & 63;
        As[r][k] = (__bf16)((k < NBRF) ? Araw[r * NBRF + k] : 0.0f);
    }
    int wave = tid >> 5, lane = tid & 31;
    int h = lane >> 4, nn = lane & 15;

    for (int l = 0; l < NCONV; ++l) {
        __syncthreads();  // As convert done / previous layer's Bs,hR readers done
        for (int i = tid; i < 64 * 48; i += 128) {
            int k = i / 48, n = i % 48;
            float v = (k < NBRF && n < NBRF) ? rW1[(l * NBRF + k) * NBRF + n] : 0.0f;
            Bs[k][n] = (__bf16)v;
        }
        if (tid < 48) w2col[tid] = (tid < NBRF) ? rW2[(l * NBRF + tid) * 9] : 0.0f;
        __syncthreads();

        v8f c[3] = {};
#pragma unroll
        for (int ks = 0; ks < 2; ++ks) {
            v16bf a = pack_A(&As[wave * 16][0], 64, lane, ks * 32);
#pragma unroll
            for (int nt = 0; nt < 3; ++nt) {
                v16bf b = pack_B(&Bs[0][nt * 16], 48, lane, ks * 32);
                c[nt]   = wmma_bf16(a, b, c[nt]);
            }
        }
#pragma unroll
        for (int nt = 0; nt < 3; ++nt) {
            int   n    = nt * 16 + nn;
            float bias = (n < NBRF) ? rb1[l * NBRF + n] : 0.0f;
#pragma unroll
            for (int r = 0; r < 8; ++r)
                hR[wave * 16 + r + 8 * h][n] = sp_f(c[nt][r] + bias);
        }
        __syncthreads();
        if (tid < 64) {  // mat-vec with rW2[:,0], fold Y0 (fixed-order dot)
            float acc = rb2[l * 9];
#pragma unroll
            for (int k = 0; k < NBRF; ++k) acc += hR[tid][k] * w2col[k];
            w0out[(long)l * EDGES + row0 + tid] = acc * kY0;
        }
    }
}

// ---------------- K3: weighted neighbor gather  s = mean_m(w0 * x[nbr]) ----------------
__global__ __launch_bounds__(256) void cg_gather_kernel(
    const float* __restrict__ x, const float* __restrict__ w0,
    const int* __restrict__ nbr_idx, float* __restrict__ s) {
    int t    = threadIdx.x;
    int atom = blockIdx.x * 16 + (t >> 4);
    int f4   = (t & 15) * 4;
    float4 acc = {0.f, 0.f, 0.f, 0.f};
#pragma unroll
    for (int m = 0; m < M_NBR; ++m) {
        int   j = nbr_idx[atom * M_NBR + m];
        float w = w0[atom * M_NBR + m];
        float4 xv = *(const float4*)&x[(long)j * FDIM + f4];
        acc.x += w * xv.x; acc.y += w * xv.y;
        acc.z += w * xv.z; acc.w += w * xv.w;
    }
    const float inv = 1.0f / (float)M_NBR;
    acc.x *= inv; acc.y *= inv; acc.z *= inv; acc.w *= inv;
    *(float4*)&s[(long)atom * FDIM + f4] = acc;
}

// ---------------- K4: y = x + (s @ tpW)*TP_NORM, + per-block BN partials ----------------
// Deterministic: each wave writes its own redsum[wave][f] slot; thread f then
// combines the 4 wave partials in fixed order into partials[block][2][64].
__global__ __launch_bounds__(128) void cg_tp_kernel(
    const float* __restrict__ s, const float* __restrict__ tpW,
    const float* __restrict__ x, float* __restrict__ y,
    float* __restrict__ partials) {
    __shared__ float  Sraw[64 * FDIM];   // raw f32 tile of s (async)
    __shared__ __bf16 As[64][64];
    __shared__ __bf16 Bs[64][64];
    __shared__ float  redsum[4][FDIM], redsq[4][FDIM];
    int tid = threadIdx.x, row0 = blockIdx.x * 64;

    // Tail block over-reads past s's end but stays inside the workspace
    // allocation (y/partials follow); those rows' outputs are discarded.
    stage_async(&Sraw[0], s + (long)row0 * FDIM, (64 * FDIM) / 4, tid, 128);
    for (int i = tid; i < 64 * 64; i += 128)
        Bs[i >> 6][i & 63] = (__bf16)tpW[i];
    __syncthreads();
    for (int i = tid; i < 64 * 64; i += 128)
        As[i >> 6][i & 63] = (__bf16)Sraw[i];
    __syncthreads();

    int wave = tid >> 5, lane = tid & 31;
    int h = lane >> 4, nn = lane & 15;
    v8f c[4] = {};
#pragma unroll
    for (int ks = 0; ks < 2; ++ks) {
        v16bf a = pack_A(&As[wave * 16][0], 64, lane, ks * 32);
#pragma unroll
        for (int nt = 0; nt < 4; ++nt) {
            v16bf b = pack_B(&Bs[0][nt * 16], 64, lane, ks * 32);
            c[nt]   = wmma_bf16(a, b, c[nt]);
        }
    }
#pragma unroll
    for (int nt = 0; nt < 4; ++nt) {
        float psum = 0.0f, psq = 0.0f;
#pragma unroll
        for (int r = 0; r < 8; ++r) {
            int gm = row0 + wave * 16 + r + 8 * h;
            if (gm < NATOMS) {
                long  o = (long)gm * FDIM + nt * 16 + nn;
                float v = c[nt][r] * kTPNORM + x[o];
                y[o] = v;
                psum += v; psq += v * v;
            }
        }
        psum += __shfl_xor(psum, 16, 32);
        psq  += __shfl_xor(psq, 16, 32);
        if (h == 0) {                        // single writer per slot
            redsum[wave][nt * 16 + nn] = psum;
            redsq[wave][nt * 16 + nn]  = psq;
        }
    }
    __syncthreads();
    if (tid < FDIM) {                        // fixed combination order
        float a = ((redsum[0][tid] + redsum[1][tid]) + redsum[2][tid]) + redsum[3][tid];
        float q = ((redsq[0][tid] + redsq[1][tid]) + redsq[2][tid]) + redsq[3][tid];
        partials[(long)blockIdx.x * (2 * FDIM) + tid]        = a;
        partials[(long)blockIdx.x * (2 * FDIM) + FDIM + tid] = q;
    }
}

// ---------------- K4b: deterministic final BN-stat reduction ----------------
__global__ __launch_bounds__(128) void cg_bnstat_kernel(
    const float* __restrict__ partials, float* __restrict__ stats) {
    int t = threadIdx.x;          // t<64: sums, t>=64: sum-of-squares
    int which = t >> 6, f = t & 63;
    float acc = 0.0f;
    for (int b = 0; b < NB_TP; ++b)   // fixed sequential order
        acc += partials[(long)b * (2 * FDIM) + which * FDIM + f];
    stats[which * FDIM + f] = acc;
}

// ---------------- K5: batchnorm + softplus -> x ----------------
__global__ __launch_bounds__(256) void cg_bn_kernel(
    const float* __restrict__ y, const float* __restrict__ stats,
    const float* __restrict__ gamma, const float* __restrict__ beta,
    float* __restrict__ x) {
    long i = (long)blockIdx.x * 256 + threadIdx.x;
    if (i >= (long)NATOMS * FDIM) return;
    int   f   = (int)(i & 63);
    float mu  = stats[f] * (1.0f / NATOMS);
    float var = stats[FDIM + f] * (1.0f / NATOMS) - mu * mu;
    float v   = gamma[f] * (y[i] - mu) * rsqrtf(var + kEPS) + beta[f];
    x[i] = sp_f(v);
}

// ---------------- K6: per-crystal mean pool ----------------
__global__ __launch_bounds__(64) void cg_pool_kernel(
    const float* __restrict__ x, const int* __restrict__ cidx,
    float* __restrict__ crys) {
    int b = blockIdx.x, f = threadIdx.x;
    float acc = 0.0f;
    for (int a = 0; a < NPC; ++a) {
        int n = cidx[b * NPC + a];
        acc += x[(long)n * FDIM + f];
    }
    crys[b * FDIM + f] = acc * (1.0f / NPC);
}

// ---------------- K7: head  h = sp(crys@fcW+fcb); out = h@outW+outb ----------------
__global__ __launch_bounds__(32) void cg_head_kernel(
    const float* __restrict__ crys, const float* __restrict__ fcW,
    const float* __restrict__ fcb, const float* __restrict__ outW,
    const float* __restrict__ outb, float* __restrict__ out,
    float* __restrict__ hout) {
    __shared__ __bf16 As[16][64];
    __shared__ __bf16 Bs[64][HFDIM];
    __shared__ float  Hs[16][HFDIM];
    int lane = threadIdx.x, row0 = blockIdx.x * 16;

    for (int i = lane; i < 16 * 64; i += 32) {
        int r = i >> 6, k = i & 63, gr = row0 + r;
        As[r][k] = (__bf16)((gr < BCRY) ? crys[gr * FDIM + k] : 0.0f);
    }
    for (int i = lane; i < 64 * HFDIM; i += 32)
        Bs[i >> 7][i & 127] = (__bf16)fcW[i];
    __syncthreads();

    v8f c[8] = {};
#pragma unroll
    for (int ks = 0; ks < 2; ++ks) {
        v16bf a = pack_A(&As[0][0], 64, lane, ks * 32);
#pragma unroll
        for (int nt = 0; nt < 8; ++nt) {
            v16bf b = pack_B(&Bs[0][nt * 16], HFDIM, lane, ks * 32);
            c[nt]   = wmma_bf16(a, b, c[nt]);
        }
    }
    int h = lane >> 4, nn = lane & 15;
#pragma unroll
    for (int nt = 0; nt < 8; ++nt) {
        float bias = fcb[nt * 16 + nn];
#pragma unroll
        for (int r = 0; r < 8; ++r) {
            int   m  = r + 8 * h, gm = row0 + m;
            float v  = sp_f(c[nt][r] + bias);
            Hs[m][nt * 16 + nn] = v;
            if (gm < BCRY) hout[gm * HFDIM + nt * 16 + nn] = v;
        }
    }
    __syncthreads();
    int row = lane >> 1, part = lane & 1;  // 2 lanes per output row
    float acc = 0.0f;
    for (int k = 0; k < 64; ++k) acc += Hs[row][part * 64 + k] * outW[part * 64 + k];
    acc += __shfl_xor(acc, 1, 32);
    if (part == 0 && row0 + row < BCRY) out[row0 + row] = acc + outb[0];
}

// ---------------- driver ----------------
extern "C" void kernel_launch(void* const* d_in, const int* in_sizes, int n_in,
                              void* d_out, int out_size, void* d_ws, size_t ws_size,
                              hipStream_t stream) {
    const float* atom_fea = (const float*)d_in[0];
    const float* nbr_fea  = (const float*)d_in[1];
    // d_in[2] = pos (unused by reference math)
    const float* emb_W    = (const float*)d_in[3];
    const float* emb_b    = (const float*)d_in[4];
    const float* rW1      = (const float*)d_in[5];
    const float* rb1      = (const float*)d_in[6];
    const float* rW2      = (const float*)d_in[7];
    const float* rb2      = (const float*)d_in[8];
    const float* tpW      = (const float*)d_in[9];
    const float* bn_gamma = (const float*)d_in[10];
    const float* bn_beta  = (const float*)d_in[11];
    const float* fc_W     = (const float*)d_in[12];
    const float* fc_b     = (const float*)d_in[13];
    const float* out_W    = (const float*)d_in[14];
    const float* out_b    = (const float*)d_in[15];
    const int*   nbr_idx  = (const int*)d_in[16];
    const int*   cidx     = (const int*)d_in[17];

    float* ws       = (float*)d_ws;
    float* x        = ws;                                   // [N,F]
    float* w0       = x + (size_t)NATOMS * FDIM;            // [3,E]
    float* s        = w0 + (size_t)NCONV * EDGES;           // [N,F]
    float* y        = s + (size_t)NATOMS * FDIM;            // [N,F]
    float* partials = y + (size_t)NATOMS * FDIM;            // [NB_TP,2,F]
    float* stats    = partials + (size_t)NB_TP * 2 * FDIM;  // [2,F]
    float* crys     = stats + 2 * FDIM;                     // [B,F]

    float* outp = (float*)d_out;       // [B,1]
    float* hp   = outp + BCRY;         // [B,HF]

    cg_embed_kernel<<<(NATOMS + 63) / 64, 128, 0, stream>>>(atom_fea, emb_W, emb_b, x);
    cg_edge_kernel<<<EDGES / 64, 128, 0, stream>>>(nbr_fea, rW1, rb1, rW2, rb2, w0);

    for (int l = 0; l < NCONV; ++l) {
        cg_gather_kernel<<<NATOMS / 16, 256, 0, stream>>>(x, w0 + (size_t)l * EDGES,
                                                          nbr_idx, s);
        cg_tp_kernel<<<NB_TP, 128, 0, stream>>>(
            s, tpW + (size_t)l * FDIM * FDIM, x, y, partials);
        cg_bnstat_kernel<<<1, 128, 0, stream>>>(partials, stats);
        cg_bn_kernel<<<(NATOMS * FDIM) / 256, 256, 0, stream>>>(
            y, stats, bn_gamma + l * FDIM, bn_beta + l * FDIM, x);
    }

    cg_pool_kernel<<<BCRY, 64, 0, stream>>>(x, cidx, crys);
    cg_head_kernel<<<(BCRY + 15) / 16, 32, 0, stream>>>(crys, fc_W, fc_b, out_W, out_b,
                                                        outp, hp);
}
